// Model_39427799777394
// MI455X (gfx1250) — matrix operand order repaired and checked by reading, not compiled, not run on hardware
//
#include <hip/hip_runtime.h>

#define NB 4
#define NN 2048
#define NT 24
#define NK 3
#define NFC 64
#define NFT 64
#define NT2 12
#define NPRED 24

typedef __attribute__((ext_vector_type(16))) __bf16 v16bf;
typedef __attribute__((ext_vector_type(8)))  float  v8f;
typedef __attribute__((ext_vector_type(4)))  int    v4i;

union Frag16 { v16bf v; uint4 q[2]; };
union Acc8   { v8f v; float f[8]; };

__device__ __forceinline__ unsigned short f2bf(float x) {
  union { float f; unsigned u; } v; v.f = x;
  unsigned r = v.u + 0x7fffu + ((v.u >> 16) & 1u);   // round-to-nearest-even
  return (unsigned short)(r >> 16);
}
__device__ __forceinline__ float sigm(float x) { return 1.0f / (1.0f + __expf(-x)); }

// Use CDNA5 async global->LDS DMA when the toolchain exposes it.
// Probe-learned signature: param0 = v4i addrspace(1)* (global src, non-const),
// param1 = v4i addrspace(3)* (LDS dst), then imm offset, imm cpol.
#if defined(__HIP_DEVICE_COMPILE__) && \
    __has_builtin(__builtin_amdgcn_global_load_async_to_lds_b128) && \
    __has_builtin(__builtin_amdgcn_s_wait_asynccnt)
#define USE_ASYNC_LDS 1
typedef __attribute__((address_space(1))) v4i* gp_v4i;
typedef __attribute__((address_space(3))) v4i* lp_v4i;
#endif

// ---------------------------------------------------------------------------
// Batched WMMA GEMM: C[batch](MxNcol,f32) = A[batch](MxKd,bf16) @ B[batch](KdxNcol,bf16)
// Row-major everywhere. M % 128 == 0, Kd % 32 == 0, Ncol % 8 == 0 (guards for Ncol < 64).
// Workgroup tile 128x64, 8 waves, each wave a 32x32 tile via 2x2 v_wmma_f32_16x16x32_bf16.
// A tile staged via GLOBAL_LOAD_ASYNC_TO_LDS_B128 (ASYNCcnt); B tile staged transposed.
// ---------------------------------------------------------------------------
#define BM 128
#define BN 64
#define BK 32

__global__ __launch_bounds__(256)
void wmma_gemm_bf16(const unsigned short* __restrict__ A,
                    const unsigned short* __restrict__ Bm,
                    float* __restrict__ C,
                    int M, int Ncol, int Kd,
                    long long sA, long long sB, long long sC)
{
  __shared__ __align__(16) unsigned short shA[BM * BK];   // [m][k]
  __shared__ __align__(16) unsigned short shBt[BN * BK];  // [n][k] (transposed)

  const int batch = blockIdx.z;
  const unsigned short* Ab = A  + (size_t)batch * (size_t)sA;
  const unsigned short* Bb = Bm + (size_t)batch * (size_t)sB;
  float*                Cb = C  + (size_t)batch * (size_t)sC;

  const int m0   = blockIdx.y * BM;
  const int n0   = blockIdx.x * BN;
  const int tid  = threadIdx.x;
  const int lane = tid & 31;
  const int wave = tid >> 5;     // 0..7
  const int wm   = wave >> 1;    // 0..3 (M)
  const int wn   = wave & 1;     // 0..1 (N)
  const int lrow = lane & 15;
  const int half = lane >> 4;

  Acc8 acc[2][2];
  #pragma unroll
  for (int i = 0; i < 2; ++i)
    #pragma unroll
    for (int j = 0; j < 2; ++j)
      #pragma unroll
      for (int e = 0; e < 8; ++e) acc[i][j].f[e] = 0.0f;

  for (int k0 = 0; k0 < Kd; k0 += BK) {
    // ---- stage A tile (128x32): 512 chunks of 8 bf16, 2 per thread
#if defined(USE_ASYNC_LDS)
    {
      int c = tid;
      #pragma unroll
      for (int it = 0; it < 2; ++it) {
        int row = c >> 2;
        int col = (c & 3) << 3;
        __builtin_amdgcn_global_load_async_to_lds_b128(
            (gp_v4i)(Ab + (size_t)(m0 + row) * Kd + k0 + col),
            (lp_v4i)(&shA[row * BK + col]), 0, 0);
        c += 256;
      }
    }
#else
    {
      int c = tid;
      #pragma unroll
      for (int it = 0; it < 2; ++it) {
        int row = c >> 2;
        int col = (c & 3) << 3;
        uint4 val = *(const uint4*)(Ab + (size_t)(m0 + row) * Kd + k0 + col);
        *(uint4*)(&shA[row * BK + col]) = val;
        c += 256;
      }
    }
#endif
    // ---- stage B tile (32x64) transposed into shBt[n][k]
    {
      int row  = tid >> 3;          // k-row in tile
      int colc = (tid & 7) << 3;    // col chunk base
      int gcol = n0 + colc;
      uint4 val = make_uint4(0u, 0u, 0u, 0u);
      if (gcol < Ncol)
        val = *(const uint4*)(Bb + (size_t)(k0 + row) * Ncol + gcol);
      const unsigned short* sv = (const unsigned short*)&val;
      #pragma unroll
      for (int j = 0; j < 8; ++j)
        shBt[(colc + j) * BK + row] = sv[j];
    }
    if (k0 + BK < Kd)   // pull next B tile toward the WGP
      __builtin_prefetch(Bb + (size_t)(k0 + BK + (tid >> 3)) * Ncol + n0, 0, 3);
#if defined(USE_ASYNC_LDS)
    __builtin_amdgcn_s_wait_asynccnt(0);
#endif
    __syncthreads();

    // ---- load fragments (ISA wave32 16-bit layouts)
    Frag16 aF[2], bF[2];
    #pragma unroll
    for (int fi = 0; fi < 2; ++fi) {
      int arow = wm * 32 + fi * 16 + lrow;
      // A: e=0..7 -> K = 8*half + e ; e=8..15 -> K = 16 + 8*half + (e-8)
      aF[fi].q[0] = *(const uint4*)(&shA[arow * BK + half * 8]);
      aF[fi].q[1] = *(const uint4*)(&shA[arow * BK + 16 + half * 8]);
      int brow = wn * 32 + fi * 16 + lrow;
      // B: e=0..7 -> K = 8*half + e ; e=8..15 -> K = 16 + 8*half + (e-8)
      bF[fi].q[0] = *(const uint4*)(&shBt[brow * BK + half * 8]);
      bF[fi].q[1] = *(const uint4*)(&shBt[brow * BK + 16 + half * 8]);
    }
#if defined(__HIP_DEVICE_COMPILE__)
    #pragma unroll
    for (int mi = 0; mi < 2; ++mi)
      #pragma unroll
      for (int ni = 0; ni < 2; ++ni)
        acc[mi][ni].v = __builtin_amdgcn_wmma_f32_16x16x32_bf16(
            false, aF[mi].v, false, bF[ni].v, (short)0, acc[mi][ni].v, false, false);
#endif
    __syncthreads();
  }

  // ---- store C (C layout: M = e + 8*half, N = lane&15)
  #pragma unroll
  for (int mi = 0; mi < 2; ++mi) {
    #pragma unroll
    for (int ni = 0; ni < 2; ++ni) {
      int col = n0 + wn * 32 + ni * 16 + lrow;
      if (col < Ncol) {
        #pragma unroll
        for (int e = 0; e < 8; ++e) {
          int row = m0 + wm * 32 + mi * 16 + e + half * 8;
          Cb[(size_t)row * Ncol + col] = acc[mi][ni].f[e];
        }
      }
    }
  }
}

// ---------------------------------------------------------------------------
// Elementwise / small kernels
// ---------------------------------------------------------------------------
__global__ void k_embed(const float* __restrict__ x, float* __restrict__ X0) {
  int i = blockIdx.x * blockDim.x + threadIdx.x;
  if (i >= NB * NT * NN) return;
  int n = i % NN; int t = (i / NN) % NT; int b = i / (NN * NT);
  X0[((size_t)b * NN + n) * NT + t] = x[i];
}

__global__ void k_tobf(const float* __restrict__ src, unsigned short* __restrict__ dst,
                       long long ntot) {
  long long i = (long long)blockIdx.x * blockDim.x + threadIdx.x;
  if (i < ntot) dst[i] = f2bf(src[i]);
}

// lhs[b,t,f] = sum_n X[b,n,f,t] * U1[n]   (one block per output, reduce over N)
__global__ void k_red_u1(const float* __restrict__ X, const float* __restrict__ U1,
                         float* __restrict__ lhs, int F, int Tt) {
  int o = blockIdx.x;
  int f = o % F; int t = (o / F) % Tt; int b = o / (F * Tt);
  __shared__ float red[256];
  float s = 0.0f;
  for (int n = threadIdx.x; n < NN; n += 256)
    s += X[(((size_t)b * NN + n) * F + f) * Tt + t] * U1[n];
  red[threadIdx.x] = s; __syncthreads();
  for (int st = 128; st > 0; st >>= 1) {
    if (threadIdx.x < st) red[threadIdx.x] += red[threadIdx.x + st];
    __syncthreads();
  }
  if (threadIdx.x == 0) lhs[o] = red[0];
}

// lhs2[b,t,n] = sum_f lhs[b,t,f] * U2[f,n]
__global__ void k_lhs2(const float* __restrict__ lhs, const float* __restrict__ U2,
                       float* __restrict__ lhs2, int F, int Tt) {
  int i = blockIdx.x * 256 + threadIdx.x;
  if (i >= NB * Tt * NN) return;
  int n = i % NN; int t = (i / NN) % Tt; int b = i / (NN * Tt);
  float s = 0.0f;
  for (int f = 0; f < F; ++f) s += lhs[(b * Tt + t) * F + f] * U2[(size_t)f * NN + n];
  lhs2[i] = s;
}

// rhsT[b,n,t] = sum_f U3[f] * X[b,n,f,t]
__global__ void k_rhsT(const float* __restrict__ X, const float* __restrict__ U3,
                       float* __restrict__ rhsT, int F, int Tt) {
  int i = blockIdx.x * 256 + threadIdx.x;
  if (i >= NB * NN * Tt) return;
  int t = i % Tt; int n = (i / Tt) % NN; int b = i / (Tt * NN);
  float s = 0.0f;
  for (int f = 0; f < F; ++f) s += U3[f] * X[(((size_t)b * NN + n) * F + f) * Tt + t];
  rhsT[i] = s;
}

// prod[b,i,j] = sum_n lhs2[b,i,n] * rhsT[b,n,j]
__global__ void k_prod_t(const float* __restrict__ lhs2, const float* __restrict__ rhsT,
                         float* __restrict__ prod, int Tt) {
  int o = blockIdx.x;
  int j = o % Tt; int ii = (o / Tt) % Tt; int b = o / (Tt * Tt);
  __shared__ float red[256];
  float s = 0.0f;
  for (int n = threadIdx.x; n < NN; n += 256)
    s += lhs2[((size_t)b * Tt + ii) * NN + n] * rhsT[((size_t)b * NN + n) * Tt + j];
  red[threadIdx.x] = s; __syncthreads();
  for (int st = 128; st > 0; st >>= 1) {
    if (threadIdx.x < st) red[threadIdx.x] += red[threadIdx.x + st];
    __syncthreads();
  }
  if (threadIdx.x == 0) prod[o] = red[0];
}

// E[b,i,t] = sum_j Ve[i,j] * sigmoid(prod[b,j,t] + be[j,t])
__global__ void k_E(const float* __restrict__ prod, const float* __restrict__ Ve,
                    const float* __restrict__ be, float* __restrict__ E, int Tt) {
  int i = blockIdx.x * 256 + threadIdx.x;
  if (i >= NB * Tt * Tt) return;
  int t = i % Tt; int ii = (i / Tt) % Tt; int b = i / (Tt * Tt);
  float s = 0.0f;
  for (int j = 0; j < Tt; ++j)
    s += Ve[ii * Tt + j] * sigm(prod[(b * Tt + j) * Tt + t] + be[j * Tt + t]);
  E[i] = s;
}

// softmax over axis 1 (i) of (B,Tt,Tt) for fixed (b,t)
__global__ void k_softmax_T(const float* __restrict__ E, float* __restrict__ tat, int Tt) {
  int i = blockIdx.x * blockDim.x + threadIdx.x;
  if (i >= NB * Tt) return;
  int t = i % Tt; int b = i / Tt;
  float mx = -1e30f;
  for (int r = 0; r < Tt; ++r) mx = fmaxf(mx, E[(b * Tt + r) * Tt + t]);
  float sm = 0.0f;
  for (int r = 0; r < Tt; ++r) sm += __expf(E[(b * Tt + r) * Tt + t] - mx);
  float inv = 1.0f / sm;
  for (int r = 0; r < Tt; ++r)
    tat[(b * Tt + r) * Tt + t] = __expf(E[(b * Tt + r) * Tt + t] - mx) * inv;
}

// x_tat[b,n,f,t] = sum_{t2} X[b,n,f,t2] * tat[b,t2,t]
__global__ void k_xtat(const float* __restrict__ X, const float* __restrict__ tat,
                       float* __restrict__ XT, int F, int Tt) {
  long long i = (long long)blockIdx.x * 256 + threadIdx.x;
  long long tot = (long long)NB * NN * F * Tt;
  if (i >= tot) return;
  int t = (int)(i % Tt); long long r = i / Tt;
  int f = (int)(r % F);  long long r2 = r / F;
  int n = (int)(r2 % NN); int b = (int)(r2 / NN);
  const float* xr = &X[(((size_t)b * NN + n) * F + f) * Tt];
  float s = 0.0f;
  for (int t2 = 0; t2 < Tt; ++t2) s += xr[t2] * tat[(b * Tt + t2) * Tt + t];
  XT[i] = s;
}

// slhs[b,n,f] = sum_t XT[b,n,f,t] * W1[t]
__global__ void k_slhs(const float* __restrict__ XT, const float* __restrict__ W1,
                       float* __restrict__ slhs, int F, int Tt) {
  long long i = (long long)blockIdx.x * 256 + threadIdx.x;
  if (i >= (long long)NB * NN * F) return;
  int f = (int)(i % F); int n = (int)((i / F) % NN); int b = (int)(i / ((long long)F * NN));
  const float* xr = &XT[(((size_t)b * NN + n) * F + f) * Tt];
  float s = 0.0f;
  for (int t = 0; t < Tt; ++t) s += xr[t] * W1[t];
  slhs[i] = s;
}

// slhs2[b,n,t] = sum_f slhs[b,n,f] * W2[f,t]
__global__ void k_slhs2(const float* __restrict__ slhs, const float* __restrict__ W2,
                        float* __restrict__ slhs2, int F, int Tt) {
  long long i = (long long)blockIdx.x * 256 + threadIdx.x;
  if (i >= (long long)NB * NN * Tt) return;
  int t = (int)(i % Tt); int n = (int)((i / Tt) % NN); int b = (int)(i / ((long long)Tt * NN));
  float s = 0.0f;
  for (int f = 0; f < F; ++f) s += slhs[((size_t)b * NN + n) * F + f] * W2[f * Tt + t];
  slhs2[i] = s;
}

// srhs[b,t,n] = sum_f W3[f] * XT[b,n,f,t]
__global__ void k_srhs(const float* __restrict__ XT, const float* __restrict__ W3,
                       float* __restrict__ srhs, int F, int Tt) {
  long long i = (long long)blockIdx.x * 256 + threadIdx.x;
  if (i >= (long long)NB * NN * Tt) return;
  int t = (int)(i % Tt); int n = (int)((i / Tt) % NN); int b = (int)(i / ((long long)Tt * NN));
  float s = 0.0f;
  for (int f = 0; f < F; ++f) s += W3[f] * XT[(((size_t)b * NN + n) * F + f) * Tt + t];
  srhs[((size_t)b * Tt + t) * NN + n] = s;
}

// sig[b,j,n] = bf16( sigmoid( sum_t slhs2[b,j,t]*srhs[b,t,n] + bs[j,n] ) )
__global__ void k_sig(const float* __restrict__ slhs2, const float* __restrict__ srhs,
                      const float* __restrict__ bs, unsigned short* __restrict__ sig, int Tt) {
  long long i = (long long)blockIdx.x * 256 + threadIdx.x;
  if (i >= (long long)NB * NN * NN) return;
  int n = (int)(i % NN); int j = (int)((i / NN) % NN); int b = (int)(i / ((long long)NN * NN));
  float s = 0.0f;
  for (int t = 0; t < Tt; ++t)
    s += slhs2[((size_t)b * NN + j) * Tt + t] * srhs[((size_t)b * Tt + t) * NN + n];
  s += bs[(size_t)j * NN + n];
  sig[i] = f2bf(sigm(s));
}

// in-place softmax over axis 1 (rows i) of (B,NN,NN); one thread per column
__global__ void k_softmax_N(float* __restrict__ S) {
  int n = blockIdx.x * 256 + threadIdx.x;
  int b = blockIdx.y;
  float* Sb = S + (size_t)b * NN * NN;
  float mx = -1e30f;
  for (int r = 0; r < NN; ++r) mx = fmaxf(mx, Sb[(size_t)r * NN + n]);
  float sm = 0.0f;
  for (int r = 0; r < NN; ++r) sm += __expf(Sb[(size_t)r * NN + n] - mx);
  float inv = 1.0f / sm;
  for (int r = 0; r < NN; ++r) {
    float v = __expf(Sb[(size_t)r * NN + n] - mx) * inv;
    Sb[(size_t)r * NN + n] = v;
  }
}

// Ak[b,n,m] = bf16( cheb_k[m,n] * sat[b,m,n] )   (transposed mask for GEMM)
__global__ void k_chebmask(const float* __restrict__ cheb_k, const float* __restrict__ sat,
                           unsigned short* __restrict__ Ak) {
  long long i = (long long)blockIdx.x * 256 + threadIdx.x;
  if (i >= (long long)NB * NN * NN) return;
  int n = (int)(i % NN); int m = (int)((i / NN) % NN); int b = (int)(i / ((long long)NN * NN));
  float v = cheb_k[(size_t)m * NN + n] * sat[((size_t)b * NN + m) * NN + n];
  Ak[((size_t)b * NN + n) * NN + m] = f2bf(v);
}

// g[b,n,o,t] (+)= sum_f tmp[b,n,f,t] * Theta_k[f,o]
__global__ void k_theta(const float* __restrict__ tmp, const float* __restrict__ Theta_k,
                        float* __restrict__ g, int F, int Tt, int initFlag) {
  long long i = (long long)blockIdx.x * 256 + threadIdx.x;
  long long tot = (long long)NB * NN * NFC * Tt;
  if (i >= tot) return;
  int t = (int)(i % Tt); int o = (int)((i / Tt) % NFC);
  long long r = i / ((long long)Tt * NFC);
  int n = (int)(r % NN); int b = (int)(r / NN);
  float s = initFlag ? 0.0f : g[i];
  for (int f = 0; f < F; ++f)
    s += tmp[(((size_t)b * NN + n) * F + f) * Tt + t] * Theta_k[f * NFC + o];
  g[i] = s;
}

// fused: tc = conv1x3(relu(g)), res = conv1x1(x), h = relu(res+tc), layernorm over FT
__global__ __launch_bounds__(64)
void k_out(const float* __restrict__ g, const float* __restrict__ X,
           const float* __restrict__ tw, const float* __restrict__ tb,
           const float* __restrict__ rw, const float* __restrict__ rb,
           const float* __restrict__ lng, const float* __restrict__ lnb,
           float* __restrict__ Xn, int F, int Tt, int stride, int Tout) {
  int tp = blockIdx.x, n = blockIdx.y, b = blockIdx.z, o = threadIdx.x;
  __shared__ float red[64];
  float tc = tb[o];
  const float* gr = &g[((size_t)b * NN + n) * NFC * Tt];
  for (int fc = 0; fc < NFC; ++fc) {
    const float* grow = gr + fc * Tt;
    #pragma unroll
    for (int r = 0; r < 3; ++r) {
      int tpos = tp * stride + r - 1;
      if (tpos >= 0 && tpos < Tt)
        tc += fmaxf(grow[tpos], 0.0f) * tw[(o * NFC + fc) * 3 + r];
    }
  }
  float res = rb[o];
  const float* xr = &X[((size_t)b * NN + n) * F * Tt];
  for (int f = 0; f < F; ++f) res += xr[f * Tt + tp * stride] * rw[o * F + f];
  float hv = fmaxf(res + tc, 0.0f);
  red[o] = hv; __syncthreads();
  for (int st = 32; st > 0; st >>= 1) { if (o < st) red[o] += red[o + st]; __syncthreads(); }
  float mu = red[0] * (1.0f / NFT); __syncthreads();
  red[o] = (hv - mu) * (hv - mu); __syncthreads();
  for (int st = 32; st > 0; st >>= 1) { if (o < st) red[o] += red[o + st]; __syncthreads(); }
  float var = red[0] * (1.0f / NFT);
  float xn = lng[o] * (hv - mu) * rsqrtf(var + 1e-5f) + lnb[o];
  Xn[(((size_t)b * NN + n) * NFT + o) * Tout + tp] = xn;
}

// out[b,p,n] = sum_{t,f} X2[b,n,f,t] * fw[p,t,f] + fb[p]
__global__ void k_final(const float* __restrict__ X2, const float* __restrict__ fw,
                        const float* __restrict__ fb, float* __restrict__ out) {
  int i = blockIdx.x * 256 + threadIdx.x;
  if (i >= NB * NPRED * NN) return;
  int n = i % NN; int p = (i / NN) % NPRED; int b = i / (NN * NPRED);
  const float* xr = &X2[((size_t)b * NN + n) * NFT * NT2];
  const float* wr = &fw[p * NT2 * NFT];
  float s = fb[p];
  for (int t = 0; t < NT2; ++t)
    for (int f = 0; f < NFT; ++f) s += xr[f * NT2 + t] * wr[t * NFT + f];
  out[i] = s;
}

// ---------------------------------------------------------------------------
// Host orchestration
// ---------------------------------------------------------------------------
static inline int cdiv(long long a, long long b) { return (int)((a + b - 1) / b); }

struct BlockWS {
  float *XT, *G, *S, *lhs, *lhs2, *rhsT, *prod, *E, *tat, *slhs, *slhs2, *srhs;
  unsigned short *sig, *vsb, *xbf;
};

static void run_block(const float* X, float* Xn, void* const* P, const float* cheb,
                      int F, int Tt, int stride, int Tout, const BlockWS& w, hipStream_t s)
{
  const float* U1 = (const float*)P[0];  const float* U2 = (const float*)P[1];
  const float* U3 = (const float*)P[2];  const float* be = (const float*)P[3];
  const float* Ve = (const float*)P[4];  const float* W1 = (const float*)P[5];
  const float* W2 = (const float*)P[6];  const float* W3 = (const float*)P[7];
  const float* bs = (const float*)P[8];  const float* Vs = (const float*)P[9];
  const float* Th = (const float*)P[10];
  const float* tw = (const float*)P[11]; const float* tb = (const float*)P[12];
  const float* rw = (const float*)P[13]; const float* rb = (const float*)P[14];
  const float* lng = (const float*)P[15]; const float* lnb = (const float*)P[16];

  // temporal attention
  k_red_u1<<<NB * Tt * F, 256, 0, s>>>(X, U1, w.lhs, F, Tt);
  k_lhs2<<<cdiv((long long)NB * Tt * NN, 256), 256, 0, s>>>(w.lhs, U2, w.lhs2, F, Tt);
  k_rhsT<<<cdiv((long long)NB * NN * Tt, 256), 256, 0, s>>>(X, U3, w.rhsT, F, Tt);
  k_prod_t<<<NB * Tt * Tt, 256, 0, s>>>(w.lhs2, w.rhsT, w.prod, Tt);
  k_E<<<cdiv((long long)NB * Tt * Tt, 256), 256, 0, s>>>(w.prod, Ve, be, w.E, Tt);
  k_softmax_T<<<cdiv(NB * Tt, 64), 64, 0, s>>>(w.E, w.tat, Tt);
  long long nx = (long long)NB * NN * F * Tt;
  k_xtat<<<cdiv(nx, 256), 256, 0, s>>>(X, w.tat, w.XT, F, Tt);

  // spatial attention
  k_slhs<<<cdiv((long long)NB * NN * F, 256), 256, 0, s>>>(w.XT, W1, w.slhs, F, Tt);
  k_slhs2<<<cdiv((long long)NB * NN * Tt, 256), 256, 0, s>>>(w.slhs, W2, w.slhs2, F, Tt);
  k_srhs<<<cdiv((long long)NB * NN * Tt, 256), 256, 0, s>>>(w.XT, W3, w.srhs, F, Tt);
  long long nnn = (long long)NB * NN * NN;
  k_sig<<<cdiv(nnn, 256), 256, 0, s>>>(w.slhs2, w.srhs, bs, w.sig, Tt);
  k_tobf<<<cdiv((long long)NN * NN, 256), 256, 0, s>>>(Vs, w.vsb, (long long)NN * NN);
  dim3 g1(NN / BN, NN / BM, NB);
  wmma_gemm_bf16<<<g1, 256, 0, s>>>(w.vsb, w.sig, w.S, NN, NN, NN,
                                    0, (long long)NN * NN, (long long)NN * NN);
  k_softmax_N<<<dim3(NN / 256, NB), 256, 0, s>>>(w.S);

  // chebyshev graph conv (A_k reuses sig buffer, tmp reuses XT buffer)
  int FTt = F * Tt;
  k_tobf<<<cdiv(nx, 256), 256, 0, s>>>(X, w.xbf, nx);
  for (int k = 0; k < NK; ++k) {
    k_chebmask<<<cdiv(nnn, 256), 256, 0, s>>>(cheb + (size_t)k * NN * NN, w.S, w.sig);
    dim3 g2(cdiv(FTt, BN), NN / BM, NB);
    wmma_gemm_bf16<<<g2, 256, 0, s>>>(w.sig, w.xbf, w.XT, NN, FTt, NN,
                                      (long long)NN * NN, (long long)NN * FTt,
                                      (long long)NN * FTt);
    k_theta<<<cdiv((long long)NB * NN * NFC * Tt, 256), 256, 0, s>>>(
        w.XT, Th + (size_t)k * F * NFC, w.G, F, Tt, (k == 0) ? 1 : 0);
  }

  // fused tconv + rconv + relu + layernorm
  k_out<<<dim3(Tout, NN, NB), NFT, 0, s>>>(w.G, X, tw, tb, rw, rb, lng, lnb,
                                           Xn, F, Tt, stride, Tout);
}

extern "C" void kernel_launch(void* const* d_in, const int* in_sizes, int n_in,
                              void* d_out, int out_size, void* d_ws, size_t ws_size,
                              hipStream_t stream)
{
  (void)in_sizes; (void)n_in; (void)out_size; (void)ws_size;
  // insertion-order flattening of setup_inputs():
  // 0:x  1..3:marks  4..20:block0 params  21..37:block1 params  38:final_w 39:final_b 40:cheb
  const float* x    = (const float*)d_in[0];
  void* const* P0   = d_in + 4;
  void* const* P1   = d_in + 21;
  const float* fw   = (const float*)d_in[38];
  const float* fb   = (const float*)d_in[39];
  const float* cheb = (const float*)d_in[40];
  float* out = (float*)d_out;

  char* base = (char*)d_ws; size_t off = 0;
  auto alloc = [&](size_t bytes) -> char* {
    char* p = base + off; off = (off + bytes + 255) & ~(size_t)255; return p;
  };
  const size_t XMAX = (size_t)NB * NN * NFT * NT2;   // 6,291,456 elements
  float* Xa = (float*)alloc(XMAX * 4);
  float* Xb = (float*)alloc(XMAX * 4);
  BlockWS w;
  w.XT    = (float*)alloc(XMAX * 4);
  w.G     = (float*)alloc((size_t)NB * NN * NFC * NT * 4);
  w.S     = (float*)alloc((size_t)NB * NN * NN * 4);
  w.sig   = (unsigned short*)alloc((size_t)NB * NN * NN * 2);
  w.vsb   = (unsigned short*)alloc((size_t)NN * NN * 2);
  w.xbf   = (unsigned short*)alloc(XMAX * 2);
  w.lhs   = (float*)alloc((size_t)NB * NT * NFT * 4);
  w.lhs2  = (float*)alloc((size_t)NB * NT * NN * 4);
  w.rhsT  = (float*)alloc((size_t)NB * NN * NT * 4);
  w.prod  = (float*)alloc((size_t)NB * NT * NT * 4);
  w.E     = (float*)alloc((size_t)NB * NT * NT * 4);
  w.tat   = (float*)alloc((size_t)NB * NT * NT * 4);
  w.slhs  = (float*)alloc((size_t)NB * NN * NFT * 4);
  w.slhs2 = (float*)alloc((size_t)NB * NN * NT * 4);
  w.srhs  = (float*)alloc((size_t)NB * NN * NT * 4);

  k_embed<<<cdiv((long long)NB * NT * NN, 256), 256, 0, stream>>>(x, Xa);
  run_block(Xa, Xb, P0, cheb, 1,   NT,  2, NT2, w, stream);   // block0: F=1,  T=24 -> T'=12
  run_block(Xb, Xa, P1, cheb, NFC, NT2, 1, NT2, w, stream);   // block1: F=64, T=12 -> T'=12
  k_final<<<cdiv((long long)NB * NPRED * NN, 256), 256, 0, stream>>>(Xa, fw, fb, out);
}
